// Decoder_60078002536670
// MI455X (gfx1250) — matrix operand order repaired and checked by reading, not compiled, not run on hardware
//
#include <hip/hip_runtime.h>

typedef __attribute__((ext_vector_type(16))) _Float16 v16h;
typedef __attribute__((ext_vector_type(8)))  _Float16 v8h;
typedef __attribute__((ext_vector_type(8)))  float    v8f;

#define B_TOT 16384
#define E_DIM 64
#define H_DIM 128
#define G4    512      // 4*H
#define SEQ   64
#define WAVES 8
#define ROWS_PER_BLOCK 128   // 8 waves * 16 rows

// ---- LDS layout (dynamic shared) --------------------------------------
// W_ih swizzled B-fragments : f16[32768]  @ 0        (64 KB)
// W_hh swizzled B-fragments : f16[65536]  @ 65536    (128 KB)
// W_emb packed {w0,w1,b,0}  : float4[64]  @ 196608   (1 KB)
// W_pos [2][128]            : float[256]  @ 197632   (1 KB)
// bias  b_ih+b_hh [512]     : float[512]  @ 198656   (2 KB)
// b_pos [2]                 : float[2]    @ 200704
// per-wave h staging 16x136 : f16[2176]*8 @ 200720   (34 KB, padded stride)
#define WIH_OFF  0
#define WHH_OFF  65536
#define EMB_OFF  196608
#define WPOS_OFF 197632
#define BIAS_OFF 198656
#define BPOS_OFF 200704
#define HSTG_OFF 200720
#define HST_STRIDE 136            // f16 elements per staged row (128 + 8 pad)
#define HST_PER_WAVE (16*HST_STRIDE)
#define SMEM_BYTES (HSTG_OFF + WAVES*HST_PER_WAVE*2)   // 235536 bytes < 320 KB

__device__ __forceinline__ float sigm(float x) {
    return 1.0f / (1.0f + __expf(-x));
}
__device__ __forceinline__ float tanh_fast(float x) {
    float e = __expf(-2.0f * x);
    return (1.0f - e) / (1.0f + e);
}
__device__ __forceinline__ v16h cat8(v8h lo, v8h hi) {
    return __builtin_shufflevector(lo, hi, 0,1,2,3,4,5,6,7,8,9,10,11,12,13,14,15);
}
__device__ __forceinline__ v8f wmma_f16(v16h a, v16h b, v8f c) {
    return __builtin_amdgcn_wmma_f32_16x16x32_f16(false, a, false, b, (short)0, c, false, false);
}

__global__ __launch_bounds__(256, 1)
void lstm_decoder_kernel(const float* __restrict__ last_pos_rel,
                         const float* __restrict__ hh,
                         const float* __restrict__ ch,
                         const float* __restrict__ W_ih,   // [512,64] row-major
                         const float* __restrict__ W_hh,   // [512,128] row-major
                         const float* __restrict__ b_ih,
                         const float* __restrict__ b_hh,
                         const float* __restrict__ W_emb,  // [64,2]
                         const float* __restrict__ b_emb,  // [64]
                         const float* __restrict__ W_pos,  // [2,128]
                         const float* __restrict__ b_pos,  // [2]
                         float* __restrict__ traj,         // [64,16384,2]
                         float* __restrict__ hF)           // [16384,128]
{
    extern __shared__ __align__(16) char smem[];
    _Float16* wih  = (_Float16*)(smem + WIH_OFF);
    _Float16* whh  = (_Float16*)(smem + WHH_OFF);
    float4*   emb  = (float4*)  (smem + EMB_OFF);
    float*    wpos = (float*)   (smem + WPOS_OFF);
    float*    biasl= (float*)   (smem + BIAS_OFF);
    float*    bposl= (float*)   (smem + BPOS_OFF);

    const int tid  = threadIdx.x;
    const int lane = tid & 31;
    const int wv   = tid >> 5;
    const int row0g = blockIdx.x * ROWS_PER_BLOCK + wv * 16;   // first batch row of this wave

    _Float16* hst = (_Float16*)(smem + HSTG_OFF) + wv * HST_PER_WAVE;

    // ---------------- one-time staging ---------------------------------
    // W_ih -> B-fragment order: frag fi = ntile*2 + kblk, lane l holds
//   col n = ntile*16 + (l&15), K = kblk*32 + (l>>4)*8 + (e<8 ? e : e+8)  (e = 0..15)
    for (int idx = tid; idx < G4 * E_DIM; idx += 256) {
        int fi = idx >> 9, l = (idx >> 4) & 31, e = idx & 15;
        int ntile = fi >> 1, kblk = fi & 1;
        int n = ntile * 16 + (l & 15);
        int k = kblk * 32 + ((l >> 4) << 3) + (e < 8 ? e : e + 8);
        wih[idx] = (_Float16)W_ih[n * E_DIM + k];
    }
    // W_hh -> frag fi = ntile*4 + kblk
    for (int idx = tid; idx < G4 * H_DIM; idx += 256) {
        int fi = idx >> 9, l = (idx >> 4) & 31, e = idx & 15;
        int ntile = fi >> 2, kblk = fi & 3;
        int n = ntile * 16 + (l & 15);
        int k = kblk * 32 + ((l >> 4) << 3) + (e < 8 ? e : e + 8);
        whh[idx] = (_Float16)W_hh[n * H_DIM + k];
    }
    if (tid < E_DIM) emb[tid] = make_float4(W_emb[tid*2], W_emb[tid*2+1], b_emb[tid], 0.0f);
    wpos[tid] = W_pos[tid];                                // 256 floats exactly
    for (int n = tid; n < G4; n += 256) biasl[n] = b_ih[n] + b_hh[n];
    if (tid < 2) bposl[tid] = b_pos[tid];

    // h0 -> per-wave staging (coalesced)
    for (int i = lane; i < 16 * H_DIM; i += 32) {
        int row = i >> 7, col = i & 127;
        hst[row * HST_STRIDE + col] = (_Float16)hh[(size_t)(row0g + row) * H_DIM + col];
    }
    // c0 -> registers in WMMA C/D layout: VGPR v, lane l -> row v+8*(l>>4), col t*16+(l&15)
    v8f cst[8];
    #pragma unroll
    for (int t = 0; t < 8; ++t)
        #pragma unroll
        for (int v = 0; v < 8; ++v) {
            int row = v + ((lane >> 4) << 3);
            cst[t][v] = ch[(size_t)(row0g + row) * H_DIM + t * 16 + (lane & 15)];
        }

    float prev_rel = last_pos_rel[(size_t)(row0g + (lane >> 1)) * 2 + (lane & 1)];

    __syncthreads();

    // ---------------- recurrent loop ------------------------------------
    #pragma unroll 1
    for (int s = 0; s < SEQ; ++s) {
        // rel of my row, from the lanes that computed it (r = lane>>1, j = lane&1)
        float relA = __shfl(prev_rel, 2 * (lane & 15), 32);
        float relB = __shfl(prev_rel, 2 * (lane & 15) + 1, 32);

        // x = rel @ W_emb^T + b_emb, built directly as A-fragments (16x32 f16)
        v16h xf[2];
        #pragma unroll
        for (int kb = 0; kb < 2; ++kb) {
            int base = kb * 32 + ((lane >> 4) << 3);
            #pragma unroll
            for (int i = 0; i < 16; ++i) {
                int e = base + (i < 8 ? i : i + 8);
                float4 w = emb[e];
                xf[kb][i] = (_Float16)(relA * w.x + relB * w.y + w.z);
            }
        }
        // h A-fragments from staging (previous step's h, padded conflict-free)
        v16h hf[4];
        #pragma unroll
        for (int kb = 0; kb < 4; ++kb) {
            int kbase = kb * 32 + ((lane >> 4) << 3);
            const v8h* p = (const v8h*)(hst + (lane & 15) * HST_STRIDE + kbase);
            hf[kb] = cat8(p[0], p[2]);
        }

        // gates = x@W_ih^T + h@W_hh^T + bias, then LSTM cell, per 16-col hidden tile
        #pragma unroll
        for (int t = 0; t < 8; ++t) {
            v8f accs[4];
            #pragma unroll
            for (int q = 0; q < 4; ++q) {       // i, f, g, o
                float bb = biasl[q * H_DIM + t * 16 + (lane & 15)];
                v8f acc = {bb, bb, bb, bb, bb, bb, bb, bb};
                int nt = q * 8 + t;             // global 16-col tile in [0,32)
                #pragma unroll
                for (int kb = 0; kb < 2; ++kb) {
                    const v16h* bp = (const v16h*)(wih + (((nt*2 + kb) << 9) + (lane << 4)));
                    acc = wmma_f16(xf[kb], *bp, acc);
                }
                #pragma unroll
                for (int kb = 0; kb < 4; ++kb) {
                    const v16h* bp = (const v16h*)(whh + (((nt*4 + kb) << 9) + (lane << 4)));
                    acc = wmma_f16(hf[kb], *bp, acc);
                }
                accs[q] = acc;
            }
            #pragma unroll
            for (int v = 0; v < 8; ++v) {
                float iv = sigm(accs[0][v]);
                float fv = sigm(accs[1][v]);
                float gv = tanh_fast(accs[2][v]);
                float ov = sigm(accs[3][v]);
                float cv = fv * cst[t][v] + iv * gv;
                cst[t][v] = cv;
                float hv = ov * tanh_fast(cv);
                int row = v + ((lane >> 4) << 3);
                hst[row * HST_STRIDE + t * 16 + (lane & 15)] = (_Float16)hv;
                if (s == SEQ - 1)
                    hF[(size_t)(row0g + row) * H_DIM + t * 16 + (lane & 15)] = hv;
            }
        }

        // ensure cross-lane staging stores are visible within the wave
        asm volatile("s_wait_dscnt 0x0" ::: "memory");

        // rel = h @ W_pos^T + b_pos : lane (r = lane>>1, j = lane&1) does one 128-dot
        {
            int r = lane >> 1, j = lane & 1;
            float acc = bposl[j];
            const v8h*    hp = (const v8h*)(hst + r * HST_STRIDE);
            const float4* wp = (const float4*)(wpos + j * H_DIM);
            #pragma unroll
            for (int i = 0; i < 16; ++i) {
                v8h h8 = hp[i];
                float4 wa = wp[i*2], wb = wp[i*2 + 1];
                acc += (float)h8[0]*wa.x + (float)h8[1]*wa.y + (float)h8[2]*wa.z + (float)h8[3]*wa.w
                     + (float)h8[4]*wb.x + (float)h8[5]*wb.y + (float)h8[6]*wb.z + (float)h8[7]*wb.w;
            }
            traj[(size_t)s * B_TOT * 2 + (size_t)(row0g + r) * 2 + j] = acc;
            prev_rel = acc;
        }
    }
}

extern "C" void kernel_launch(void* const* d_in, const int* in_sizes, int n_in,
                              void* d_out, int out_size, void* d_ws, size_t ws_size,
                              hipStream_t stream) {
    // setup_inputs order:
    // 0 last_pos (unused: lp is dead w.r.t. outputs), 1 last_pos_rel, 2 hh, 3 ch,
    // 4 W_ih, 5 W_hh, 6 b_ih, 7 b_hh, 8 W_emb, 9 b_emb, 10 W_pos, 11 b_pos, 12 seq_start_end
    const float* last_pos_rel = (const float*)d_in[1];
    const float* hh    = (const float*)d_in[2];
    const float* ch    = (const float*)d_in[3];
    const float* W_ih  = (const float*)d_in[4];
    const float* W_hh  = (const float*)d_in[5];
    const float* b_ih  = (const float*)d_in[6];
    const float* b_hh  = (const float*)d_in[7];
    const float* W_emb = (const float*)d_in[8];
    const float* b_emb = (const float*)d_in[9];
    const float* W_pos = (const float*)d_in[10];
    const float* b_pos = (const float*)d_in[11];

    float* out  = (float*)d_out;
    float* traj = out;                               // [64,16384,2]
    float* hFp  = out + (size_t)SEQ * B_TOT * 2;     // [16384,128]

    dim3 grid(B_TOT / ROWS_PER_BLOCK);               // 128 blocks
    dim3 block(256);                                 // 8 waves (wave32)
    lstm_decoder_kernel<<<grid, block, SMEM_BYTES, stream>>>(
        last_pos_rel, hh, ch, W_ih, W_hh, b_ih, b_hh,
        W_emb, b_emb, W_pos, b_pos, traj, hFp);
}
